// FunnelAttention_14181982011462
// MI455X (gfx1250) — matrix operand order repaired and checked by reading, not compiled, not run on hardware
//
#include <hip/hip_runtime.h>

// ---------------------------------------------------------------------------
// FunnelAttention (Transformer-XL relative attention) for gfx1250 / MI455X
// B=4, S=1024, D=1024, H=16, Dh=64.  Heavy math: v_wmma_f32_16x16x32_bf16.
// K/V tiles: double-buffered global_load_async_to_lds_b128 (ASYNCcnt).
// ---------------------------------------------------------------------------

typedef __attribute__((ext_vector_type(16))) __bf16 v16bf;
typedef __attribute__((ext_vector_type(8)))  float  v8f;

union AFrag { v16bf v; unsigned int u[8]; };

#define NB  4
#define SQ  1024
#define DD  1024
#define NH  16
#define DH  64
#define NORM 0.125f
#define BIGC 1000000.0f

#define ASYNC_KV 1   // use gfx1250 async global->LDS DMA for K/V tiles

__device__ __forceinline__ unsigned short f2bf(float f) {
  unsigned int u = __float_as_uint(f);
  u += 0x7FFFu + ((u >> 16) & 1u);           // round-to-nearest-even
  return (unsigned short)(u >> 16);
}
__device__ __forceinline__ unsigned int pack_bf2(float lo, float hi) {
  return (unsigned int)f2bf(lo) | ((unsigned int)f2bf(hi) << 16);
}
__device__ __forceinline__ float bf2f(unsigned short b) {
  return __uint_as_float(((unsigned int)b) << 16);
}
__device__ __forceinline__ v8f wmma_bf16(const AFrag& a, const AFrag& b, v8f c) {
  return __builtin_amdgcn_wmma_f32_16x16x32_bf16(false, a.v, false, b.v,
                                                 (short)0, c, false, false);
}
// LDS byte offset of a __shared__ object (low 32 bits of the generic address).
__device__ __forceinline__ unsigned lds_off(const void* p) {
  return (unsigned)(unsigned long long)p;
}

// ---------------------------------------------------------------------------
// Generic bf16 GEMM:  C[M,N] = A[M,K](f32) * B[K,N](f32), 64x64 block tile,
// K-step 32, 128 threads = 4 waves, wave w -> 16-row strip, 4 N-tiles each.
// mode 0: Q-proj  -> o0=(acc+bias0)*scale, o1=(acc+bias1)*scale  (B,H,S,Dh) bf16
// mode 1: K/V-proj-> o0=acc+bias0                                 (B,H,S,Dh) bf16
// mode 2: r_head  -> o0=acc (batched over blockIdx.z)             row-major bf16
// mode 3: Wo      -> fo = acc + bias0[col] + resid[row,col]       row-major f32
// ---------------------------------------------------------------------------
__global__ __launch_bounds__(128) void gemm_bf16_fused(
    const float* __restrict__ A, const float* __restrict__ Bm,
    int M, int N, int K, int mode,
    unsigned short* __restrict__ o0, unsigned short* __restrict__ o1,
    float* __restrict__ fo,
    const float* __restrict__ bias0, const float* __restrict__ bias1,
    const float* __restrict__ resid, float scale)
{
  __shared__ unsigned int lds_a[64 * 16];   // [row][k/2]  bf16 pairs
  __shared__ unsigned int lds_bt[64 * 16];  // [n][k/2]    (B transposed)

  const int tid  = threadIdx.x;
  const int wave = tid >> 5;
  const int lane = tid & 31;
  const int g    = lane >> 4;
  const int n16  = lane & 15;

  const int m0 = blockIdx.y * 64;
  const int n0 = blockIdx.x * 64;
  if (mode == 2) {                      // batched over heads
    Bm += (size_t)blockIdx.z * K * N;
    o0 += (size_t)blockIdx.z * M * N;
  }

  v8f acc[4];
#pragma unroll
  for (int j = 0; j < 4; ++j) acc[j] = (v8f){};

  for (int k0 = 0; k0 < K; k0 += 32) {
    {   // A tile 64x32 -> bf16 LDS
      int row = tid >> 1, seg = tid & 1;
      const float* ap = A + (size_t)(m0 + row) * K + k0 + seg * 16;
#pragma unroll
      for (int i = 0; i < 4; ++i) {
        float4 f = ((const float4*)ap)[i];
        lds_a[row * 16 + seg * 8 + i * 2 + 0] = pack_bf2(f.x, f.y);
        lds_a[row * 16 + seg * 8 + i * 2 + 1] = pack_bf2(f.z, f.w);
      }
    }
    {   // B tile 32x64 -> transposed bf16 LDS  bt[n][k]
      int kr = tid >> 2, cs = (tid & 3) * 16;
      const float* bp = Bm + (size_t)(k0 + kr) * N + n0 + cs;
      unsigned short* bt16 = (unsigned short*)lds_bt;
#pragma unroll
      for (int i = 0; i < 4; ++i) {
        float4 f = ((const float4*)bp)[i];
        int c = cs + i * 4;
        bt16[(c + 0) * 32 + kr] = f2bf(f.x);
        bt16[(c + 1) * 32 + kr] = f2bf(f.y);
        bt16[(c + 2) * 32 + kr] = f2bf(f.z);
        bt16[(c + 3) * 32 + kr] = f2bf(f.w);
      }
    }
    __syncthreads();

    AFrag af;
    {
      int row = wave * 16 + n16;
#pragma unroll
      for (int j = 0; j < 8; ++j) {
        int ku = 8 * g + j;
        af.u[j] = lds_a[row * 16 + ku];
      }
    }
#pragma unroll
    for (int jn = 0; jn < 4; ++jn) {
      AFrag bf;
      int n = jn * 16 + n16;
#pragma unroll
      for (int j = 0; j < 8; ++j) bf.u[j] = lds_bt[n * 16 + 8 * g + j];
      acc[jn] = wmma_bf16(af, bf, acc[jn]);
    }
    __syncthreads();
  }

  // ---- epilogue (mode uniform across block) ----
#pragma unroll
  for (int jn = 0; jn < 4; ++jn) {
    int col = n0 + jn * 16 + n16;
#pragma unroll
    for (int r = 0; r < 8; ++r) {
      int row = m0 + wave * 16 + r + 8 * g;
      float v = acc[jn][r];
      if (mode == 0) {
        int h = col >> 6, dh = col & 63, bq = row >> 10, s = row & 1023;
        size_t idx = (((size_t)(bq * NH + h)) * SQ + s) * DH + dh;
        o0[idx] = f2bf((v + bias0[h * DH + dh]) * scale);
        o1[idx] = f2bf((v + bias1[h * DH + dh]) * scale);
      } else if (mode == 1) {
        int h = col >> 6, dh = col & 63, bq = row >> 10, s = row & 1023;
        size_t idx = (((size_t)(bq * NH + h)) * SQ + s) * DH + dh;
        o0[idx] = f2bf(v + bias0[h * DH + dh]);
      } else if (mode == 2) {
        o0[(size_t)row * 64 + col] = f2bf(v);
      } else {
        size_t idx = (size_t)row * N + col;
        fo[idx] = v + bias0[col] + resid[idx];
      }
    }
  }
}

// ---------------------------------------------------------------------------
// Segment logits: same/diff[b,h,s] = (qh + r_s_bias*norm) . seg_embed[{1,0},h,:]
// qh reconstructed from stored qh_w = (qh + r_w_bias)*norm.
// ---------------------------------------------------------------------------
__global__ __launch_bounds__(256) void seg_kernel(
    const unsigned short* __restrict__ qh_w,
    const float* __restrict__ r_w_bias, const float* __restrict__ r_s_bias,
    const float* __restrict__ seg_embed,
    float* __restrict__ same, float* __restrict__ diff)
{
  int idx = blockIdx.x * blockDim.x + threadIdx.x;      // (b*H+h)*S + s
  int h = (idx >> 10) & (NH - 1);
  const unsigned short* qr = qh_w + (size_t)idx * DH;
  float sm = 0.f, df = 0.f;
#pragma unroll 8
  for (int d = 0; d < DH; ++d) {
    float qv = bf2f(qr[d]) * (1.0f / NORM) - r_w_bias[h * DH + d]
             + r_s_bias[h * DH + d] * NORM;
    df += qv * seg_embed[(0 * NH + h) * DH + d];
    sm += qv * seg_embed[(1 * NH + h) * DH + d];
  }
  same[idx] = sm; diff[idx] = df;
}

// ---------------------------------------------------------------------------
// Fused flash attention with relative shift:
//   score(q,k) = qh_w.kh + pos(q, S+k-q) + seg +- mask, online softmax, P@V.
// Grid (S/64, H, B), 128 threads = 4 waves; wave owns 16 query rows.
// K/V tiles double-buffered via async global->LDS DMA.
// ---------------------------------------------------------------------------
__global__ __launch_bounds__(128) void attn_kernel(
    const unsigned short* __restrict__ qh_w, const unsigned short* __restrict__ qh_r,
    const unsigned short* __restrict__ kh,   const unsigned short* __restrict__ vh,
    const unsigned short* __restrict__ r_head,
    const float* __restrict__ seg_same, const float* __restrict__ seg_diff,
    const unsigned char* __restrict__ seg_mat, const float* __restrict__ attn_mask,
    float* __restrict__ attn_out)
{
  __shared__ unsigned int kv_lds[2][2][64 * 32];  // [buf][K/V][key][dh/2] bf16
  __shared__ float        pos_spill[4][16][80];   // per-wave shifted pos band
  __shared__ unsigned int pbuf[4][16][32];        // per-wave P tile (bf16)

  const int tid  = threadIdx.x;
  const int wave = tid >> 5;
  const int lane = tid & 31;
  const int g    = lane >> 4;
  const int n16  = lane & 15;

  const int q0  = blockIdx.x * 64;
  const int h   = blockIdx.y;
  const int b   = blockIdx.z;
  const int qw0 = q0 + wave * 16;

  const unsigned int* qw_u  = (const unsigned int*)qh_w + (size_t)(b * NH + h) * SQ * 32;
  const unsigned int* qr_u  = (const unsigned int*)qh_r + (size_t)(b * NH + h) * SQ * 32;
  const char* kh_g = (const char*)(kh + (size_t)(b * NH + h) * SQ * DH);
  const char* vh_g = (const char*)(vh + (size_t)(b * NH + h) * SQ * DH);
  const unsigned int* rh_u  = (const unsigned int*)r_head + (size_t)h * (2 * SQ) * 32;

  // Each wave DMAs a quarter of the 8KB K tile and 8KB V tile:
  // per asm op: 32 lanes x 16B = 512B; 4 ops per array per wave.
  const unsigned kbuf0 = lds_off(&kv_lds[0][0][0]);
  const unsigned kbuf1 = lds_off(&kv_lds[1][0][0]);
#if ASYNC_KV
  #define ISSUE_TILE(buf_base, koff)                                          \
    do {                                                                      \
      _Pragma("unroll")                                                       \
      for (int i_ = 0; i_ < 4; ++i_) {                                        \
        unsigned co_ = (((unsigned)wave * 4 + i_) * 32 + lane) * 16;          \
        unsigned long long ka_ = (unsigned long long)(kh_g + (koff) + co_);   \
        unsigned long long va_ = (unsigned long long)(vh_g + (koff) + co_);   \
        asm volatile("global_load_async_to_lds_b128 %0, %1, off"              \
                     :: "v"((buf_base) + co_), "v"(ka_) : "memory");          \
        asm volatile("global_load_async_to_lds_b128 %0, %1, off"              \
                     :: "v"((buf_base) + 8192u + co_), "v"(va_) : "memory");  \
      }                                                                       \
    } while (0)
#else
  #define ISSUE_TILE(buf_base, koff)                                          \
    do {                                                                      \
      unsigned int* kd_ = &kv_lds[((buf_base) == kbuf1) ? 1 : 0][0][0];       \
      int row_ = tid >> 1, seg_ = tid & 1;                                    \
      const uint4* ks_ = (const uint4*)(kh_g + (koff) + row_ * 128 + seg_ * 64); \
      const uint4* vs_ = (const uint4*)(vh_g + (koff) + row_ * 128 + seg_ * 64); \
      uint4* k4_ = (uint4*)(kd_ + row_ * 32 + seg_ * 16);                     \
      uint4* v4_ = (uint4*)(kd_ + 2048 + row_ * 32 + seg_ * 16);              \
      _Pragma("unroll")                                                       \
      for (int i_ = 0; i_ < 4; ++i_) { k4_[i_] = ks_[i_]; v4_[i_] = vs_[i_]; } \
    } while (0)
#endif

  // A-fragments for Q (layout matches global (s,dh) bf16 rows directly)
  AFrag aW[2], aR[2];
  {
    int row = qw0 + n16;
#pragma unroll
    for (int kk = 0; kk < 2; ++kk)
#pragma unroll
      for (int j = 0; j < 8; ++j) {
        int ku = 16 * kk + 8 * g + j;
        aW[kk].u[j] = qw_u[(size_t)row * 32 + ku];
        aR[kk].u[j] = qr_u[(size_t)row * 32 + ku];
      }
  }

  v8f o[4];
#pragma unroll
  for (int j = 0; j < 4; ++j) o[j] = (v8f){};
  float m_r[8], l_r[8], same_r[8], diff_r[8];
#pragma unroll
  for (int r = 0; r < 8; ++r) {
    m_r[r] = -3.0e38f; l_r[r] = 0.f;
    int q = qw0 + r + 8 * g;
    same_r[r] = seg_same[(size_t)(b * NH + h) * SQ + q];
    diff_r[r] = seg_diff[(size_t)(b * NH + h) * SQ + q];
  }

  // prologue: start DMA of tile 0 into buffer 0
  ISSUE_TILE(kbuf0, 0);

  for (int kt = 0; kt < SQ / 64; ++kt) {
    const int k0 = kt * 64;
    const int cur = kt & 1;
#if ASYNC_KV
    asm volatile("s_wait_asynccnt 0x0" ::: "memory");
#endif
    __syncthreads();            // tile kt resident & visible block-wide
    if (kt + 1 < SQ / 64)       // overlap: DMA next tile into other buffer
      ISSUE_TILE(cur ? kbuf0 : kbuf1, (k0 + 64) * 128);

    const unsigned int* klds = &kv_lds[cur][0][0];
    const unsigned short* v16 = (const unsigned short*)&kv_lds[cur][1][0];

    // ---- scores: (qh_w . kh^T), norm pre-folded ----
    v8f sacc[4];
#pragma unroll
    for (int jn = 0; jn < 4; ++jn) {
      AFrag b0, b1;
      int key = jn * 16 + n16;
#pragma unroll
      for (int j = 0; j < 8; ++j) {
        b0.u[j] = klds[key * 32 + 8 * g + j];
        b1.u[j] = klds[key * 32 + 16 + 8 * g + j];
      }
      v8f z = (v8f){};
      z = wmma_bf16(aW[0], b0, z);
      sacc[jn] = wmma_bf16(aW[1], b1, z);
    }

    // ---- positional band: p = S + k - q spans 79 columns for this strip ----
    const int p_start = SQ + k0 - qw0 - 15;
#pragma unroll
    for (int t = 0; t < 5; ++t) {
      int p = p_start + t * 16 + n16;
      p = (p < 0) ? 0 : ((p > 2 * SQ - 1) ? 2 * SQ - 1 : p);
      const unsigned int* rrow = rh_u + (size_t)p * 32;
      AFrag b0, b1;
#pragma unroll
      for (int j = 0; j < 8; ++j) {
        b0.u[j] = rrow[8 * g + j];
        b1.u[j] = rrow[16 + 8 * g + j];
      }
      v8f z = (v8f){};
      z = wmma_bf16(aR[0], b0, z);
      v8f pa = wmma_bf16(aR[1], b1, z);
#pragma unroll
      for (int r = 0; r < 8; ++r)
        pos_spill[wave][r + 8 * g][t * 16 + n16] = pa[r];
    }
    asm volatile("s_wait_dscnt 0x0" ::: "memory");

    // ---- combine: rel-shifted pos + segment + mask ----
#pragma unroll
    for (int jn = 0; jn < 4; ++jn) {
      int kcol = k0 + jn * 16 + n16;
      float mv = -BIGC * attn_mask[b * SQ + kcol];
#pragma unroll
      for (int r = 0; r < 8; ++r) {
        int m = r + 8 * g;
        float pv = pos_spill[wave][m][jn * 16 + n16 - m + 15];
        int q = qw0 + m;
        unsigned char sb = seg_mat[((size_t)(b * SQ + q)) * SQ + kcol];
        sacc[jn][r] += pv + mv + (sb ? same_r[r] : diff_r[r]);
      }
    }

    // ---- online softmax (rows shared by each 16-lane half) ----
#pragma unroll
    for (int r = 0; r < 8; ++r) {
      float rm = fmaxf(fmaxf(sacc[0][r], sacc[1][r]), fmaxf(sacc[2][r], sacc[3][r]));
      rm = fmaxf(rm, __shfl_xor(rm, 1));
      rm = fmaxf(rm, __shfl_xor(rm, 2));
      rm = fmaxf(rm, __shfl_xor(rm, 4));
      rm = fmaxf(rm, __shfl_xor(rm, 8));
      float mn  = fmaxf(m_r[r], rm);
      float esc = __expf(m_r[r] - mn);
      m_r[r] = mn;
      float ps = 0.f;
#pragma unroll
      for (int jn = 0; jn < 4; ++jn) {
        float e = __expf(sacc[jn][r] - mn);
        sacc[jn][r] = e; ps += e;
      }
      ps += __shfl_xor(ps, 1);
      ps += __shfl_xor(ps, 2);
      ps += __shfl_xor(ps, 4);
      ps += __shfl_xor(ps, 8);
      l_r[r] = l_r[r] * esc + ps;
#pragma unroll
      for (int jn = 0; jn < 4; ++jn) o[jn][r] *= esc;
    }

    // ---- P (C-layout) -> A-fragment layout via per-wave LDS transpose ----
    unsigned short* pb16 = (unsigned short*)&pbuf[wave][0][0];
#pragma unroll
    for (int jn = 0; jn < 4; ++jn)
#pragma unroll
      for (int r = 0; r < 8; ++r)
        pb16[(r + 8 * g) * 64 + jn * 16 + n16] = f2bf(sacc[jn][r]);
    asm volatile("s_wait_dscnt 0x0" ::: "memory");

    AFrag aP[2];
#pragma unroll
    for (int kk = 0; kk < 2; ++kk)
#pragma unroll
      for (int j = 0; j < 8; ++j) {
        int ku = 16 * kk + 8 * g + j;
        aP[kk].u[j] = pbuf[wave][n16][ku];
      }

    // ---- O += P @ V ----
#pragma unroll
    for (int jn = 0; jn < 4; ++jn) {
      AFrag b0, b1;
      int dh = jn * 16 + n16;
#pragma unroll
      for (int j = 0; j < 8; ++j) {
        int kA = 16 * g + 2 * j;
        b0.u[j] = (unsigned)v16[kA * 64 + dh] |
                  ((unsigned)v16[(kA + 1) * 64 + dh] << 16);
        b1.u[j] = (unsigned)v16[(32 + kA) * 64 + dh] |
                  ((unsigned)v16[(32 + kA + 1) * 64 + dh] << 16);
      }
      o[jn] = wmma_bf16(aP[0], b0, o[jn]);
      o[jn] = wmma_bf16(aP[1], b1, o[jn]);
    }
  }

  // ---- write O / l  as (B, S, H, Dh) f32 for the Wo GEMM ----
#pragma unroll
  for (int jn = 0; jn < 4; ++jn) {
    int dh = jn * 16 + n16;
#pragma unroll
    for (int r = 0; r < 8; ++r) {
      int q = qw0 + r + 8 * g;
      attn_out[(((size_t)(b * SQ + q)) * NH + h) * DH + dh] = o[jn][r] / l_r[r];
    }
  }
#undef ISSUE_TILE
}

// ---------------------------------------------------------------------------
// Row LayerNorm over D=1024.
// ---------------------------------------------------------------------------
__global__ __launch_bounds__(256) void ln_kernel(
    const float* __restrict__ x, const float* __restrict__ gamma,
    const float* __restrict__ beta, float* __restrict__ out)
{
  __shared__ float s1[256], s2[256];
  const int row = blockIdx.x;
  const float* xr = x + (size_t)row * DD;
  float sum = 0.f, sq = 0.f;
#pragma unroll
  for (int i = 0; i < 4; ++i) {
    float v = xr[threadIdx.x + i * 256];
    sum += v; sq += v * v;
  }
  s1[threadIdx.x] = sum; s2[threadIdx.x] = sq;
  __syncthreads();
  for (int st = 128; st > 0; st >>= 1) {
    if ((int)threadIdx.x < st) {
      s1[threadIdx.x] += s1[threadIdx.x + st];
      s2[threadIdx.x] += s2[threadIdx.x + st];
    }
    __syncthreads();
  }
  float mu  = s1[0] * (1.0f / DD);
  float var = s2[0] * (1.0f / DD) - mu * mu;
  float inv = rsqrtf(var + 1e-5f);
  float* orow = out + (size_t)row * DD;
#pragma unroll
  for (int i = 0; i < 4; ++i) {
    int c = threadIdx.x + i * 256;
    orow[c] = (xr[c] - mu) * inv * gamma[c] + beta[c];
  }
}

// ---------------------------------------------------------------------------
extern "C" void kernel_launch(void* const* d_in, const int* in_sizes, int n_in,
                              void* d_out, int out_size, void* d_ws, size_t ws_size,
                              hipStream_t stream) {
  (void)in_sizes; (void)n_in; (void)out_size; (void)ws_size;
  const float* q         = (const float*)d_in[0];
  const float* k         = (const float*)d_in[1];
  const float* v         = (const float*)d_in[2];
  const float* pos_enc   = (const float*)d_in[3];
  const float* attn_mask = (const float*)d_in[4];
  const unsigned char* seg_mat = (const unsigned char*)d_in[5];
  const float* Wq        = (const float*)d_in[6];
  const float* Wk        = (const float*)d_in[7];
  const float* bk        = (const float*)d_in[8];
  const float* Wv        = (const float*)d_in[9];
  const float* bv        = (const float*)d_in[10];
  const float* r_w_bias  = (const float*)d_in[11];
  const float* r_r_bias  = (const float*)d_in[12];
  const float* r_kernel  = (const float*)d_in[13];
  const float* r_s_bias  = (const float*)d_in[14];
  const float* seg_embed = (const float*)d_in[15];
  const float* Wo        = (const float*)d_in[16];
  const float* bo        = (const float*)d_in[17];
  const float* ln_gamma  = (const float*)d_in[18];
  const float* ln_beta   = (const float*)d_in[19];

  char* ws = (char*)d_ws;
  size_t off = 0;
  auto alloc = [&](size_t bytes) -> void* {
    void* p = ws + off;
    off = (off + bytes + 255) & ~(size_t)255;
    return p;
  };
  const size_t nBHSd = (size_t)NB * NH * SQ * DH;          // 4.19M elems
  unsigned short* qh_w   = (unsigned short*)alloc(nBHSd * 2);
  unsigned short* qh_r   = (unsigned short*)alloc(nBHSd * 2);
  unsigned short* kh_b   = (unsigned short*)alloc(nBHSd * 2);
  unsigned short* vh_b   = (unsigned short*)alloc(nBHSd * 2);
  unsigned short* r_head = (unsigned short*)alloc((size_t)NH * 2 * SQ * DH * 2);
  float* seg_same        = (float*)alloc((size_t)NB * NH * SQ * 4);
  float* seg_diff        = (float*)alloc((size_t)NB * NH * SQ * 4);
  float* attn_buf        = (float*)alloc((size_t)NB * SQ * DD * 4);
  float* x_buf           = (float*)alloc((size_t)NB * SQ * DD * 4);

  const int M = NB * SQ;        // 4096

  // Projections: qh (fused r_w/r_r bias + norm), kh+bk, vh+bv
  gemm_bf16_fused<<<dim3(16, 64, 1), 128, 0, stream>>>(
      q, Wq, M, DD, DD, 0, qh_w, qh_r, nullptr, r_w_bias, r_r_bias, nullptr, NORM);
  gemm_bf16_fused<<<dim3(16, 64, 1), 128, 0, stream>>>(
      k, Wk, M, DD, DD, 1, kh_b, nullptr, nullptr, bk, nullptr, nullptr, 1.f);
  gemm_bf16_fused<<<dim3(16, 64, 1), 128, 0, stream>>>(
      v, Wv, M, DD, DD, 1, vh_b, nullptr, nullptr, bv, nullptr, nullptr, 1.f);
  // r_head[h] = pos_enc @ r_kernel[h]   (batched over H)
  gemm_bf16_fused<<<dim3(1, 32, NH), 128, 0, stream>>>(
      pos_enc, r_kernel, 2 * SQ, DH, DD, 2, r_head, nullptr, nullptr,
      nullptr, nullptr, nullptr, 1.f);
  // segment logits
  seg_kernel<<<dim3((NB * NH * SQ) / 256), 256, 0, stream>>>(
      qh_w, r_w_bias, r_s_bias, seg_embed, seg_same, seg_diff);
  // fused flash attention with rel-shift (async double-buffered K/V)
  attn_kernel<<<dim3(SQ / 64, NH, NB), 128, 0, stream>>>(
      qh_w, qh_r, kh_b, vh_b, r_head, seg_same, seg_diff, seg_mat, attn_mask,
      attn_buf);
  // output projection + residual
  gemm_bf16_fused<<<dim3(16, 64, 1), 128, 0, stream>>>(
      attn_buf, Wo, M, DD, DD, 3, nullptr, nullptr, x_buf, bo, nullptr, q, 1.f);
  // layernorm
  ln_kernel<<<dim3(M), 256, 0, stream>>>(x_buf, ln_gamma, ln_beta, (float*)d_out);
}
